// WIP_69604239999308
// MI455X (gfx1250) — matrix operand order repaired and checked by reading, not compiled, not run on hardware
//
#include <hip/hip_runtime.h>
#include <hip/hip_bf16.h>
#include <math.h>

// ---------------- model constants ----------------
static constexpr int kD      = 128;
static constexpr int kS      = 19;
static constexpr int kNH     = 4;
static constexpr int kHD     = 32;     // kD / kNH
static constexpr int kFF     = 2048;
static constexpr int kVOCAB  = 30000;
static constexpr int kB      = 128;
static constexpr int kHIST   = 8;
static constexpr int kMAXLEN = 20;
static constexpr int kNL     = 3;

typedef __attribute__((ext_vector_type(16))) __bf16 v16bf;
typedef __attribute__((ext_vector_type(8)))  float  v8f;

// ---------------- bf16 helpers ----------------
static __device__ __forceinline__ unsigned short f32_to_bf16_bits(float f) {
  unsigned int u = __float_as_uint(f);
  u += 0x7FFFu + ((u >> 16) & 1u);      // round-to-nearest-even
  return (unsigned short)(u >> 16);
}
static __device__ __forceinline__ unsigned int pack_bf16x2(float lo, float hi) {
  return (unsigned int)f32_to_bf16_bits(lo) |
         ((unsigned int)f32_to_bf16_bits(hi) << 16);
}
union BfBits { unsigned short u; __bf16 b; };
static __device__ __forceinline__ __bf16 bf_from_bits(unsigned short s) {
  BfBits c; c.u = s; return c.b;
}

// ---------------- WMMA GEMM: C[M,N] = A[M,K] @ W[N,K]^T + bias ----------------
// 64x64 tile per 128-thread (4-wave) workgroup, K stepped by 32 through LDS,
// v_wmma_f32_16x16x32_bf16 per 16x16 sub-tile. Staging uses 128-bit global
// loads + packed 64-bit LDS stores; LDS rows padded (32 -> 36 halfwords) to
// stagger banks. Fragment packing used here (wave32; A and B share contiguous k order):
//   A (16x32, 16-bit): lane<16 -> M=lane, K=0..15; lane>=16 -> K=16..31 (ascending)
//   B (32x16, 16-bit): lane%16 = N column; lane<16 -> K=0..15 else K=16..31 (ascending)
//   C/D (16x16 f32):   lane%16 = N; VGPR r -> M = r + (lane<16 ? 0 : 8)
template<bool A_BF16, bool RELU, bool OUT_BF16>
__global__ __launch_bounds__(128)
void gemm_wmma(const void* __restrict__ A_, const float* __restrict__ W,
               const float* __restrict__ bias, void* __restrict__ C_,
               int M, int N, int K)
{
  constexpr int TM = 64, TN = 64, KC = 32;
  constexpr int LDP = KC + 4;               // padded LDS row stride (halfwords)
  __shared__ unsigned short lA[TM * LDP];   // [row][k]
  __shared__ unsigned short lB[TN * LDP];   // [col][k]

  const int tid  = threadIdx.x;
  const int lane = tid & 31;
  const int wv   = tid >> 5;               // wave 0..3 -> M strip
  const int m0   = blockIdx.y * TM;
  const int n0   = blockIdx.x * TN;

  const float*          Af = (const float*)A_;
  const unsigned short* Ab = (const unsigned short*)A_;

  v8f acc[4];
  #pragma unroll
  for (int t = 0; t < 4; t++)
    #pragma unroll
    for (int r = 0; r < 8; r++) acc[t][r] = 0.0f;

  for (int kc = 0; kc < K; kc += KC) {
    // prefetch next weight K-chunk (gfx1250 global_prefetch_b8, near scope)
    if (kc + KC < K) {
      int pn = n0 + (tid >> 1);
      int pk = kc + KC + ((tid & 1) << 4);
      if (pn < N) __builtin_prefetch(&W[(size_t)pn * K + pk], 0, 3);
    }
    // stage A tile (64x32): 4 passes, 4 elems/thread/pass, b128 load + b64 store
    #pragma unroll
    for (int i = tid * 4; i < TM * KC; i += 128 * 4) {
      int r = i >> 5, k = i & 31;
      int gm = m0 + r;
      uint2 p;
      if (A_BF16) {
        if (gm < M) p = *(const uint2*)(Ab + (size_t)gm * K + kc + k);
        else        p = make_uint2(0u, 0u);
      } else {
        float4 f = (gm < M) ? *(const float4*)(Af + (size_t)gm * K + kc + k)
                            : make_float4(0.f, 0.f, 0.f, 0.f);
        p.x = pack_bf16x2(f.x, f.y);
        p.y = pack_bf16x2(f.z, f.w);
      }
      *(uint2*)(&lA[r * LDP + k]) = p;
    }
    // stage W tile (64x32)
    #pragma unroll
    for (int i = tid * 4; i < TN * KC; i += 128 * 4) {
      int c = i >> 5, k = i & 31;
      int gn = n0 + c;
      float4 f = (gn < N) ? *(const float4*)(W + (size_t)gn * K + kc + k)
                          : make_float4(0.f, 0.f, 0.f, 0.f);
      uint2 p;
      p.x = pack_bf16x2(f.x, f.y);
      p.y = pack_bf16x2(f.z, f.w);
      *(uint2*)(&lB[c * LDP + k]) = p;
    }
    __syncthreads();

    // A fragment for this wave's 16-row strip
    v16bf aF;
    {
      int mrow = (wv << 4) + (lane & 15);
      int kb   = (lane < 16) ? 0 : 16;
      #pragma unroll
      for (int i = 0; i < 8; i++) aF[i]     = bf_from_bits(lA[mrow * LDP + kb + i]);
      #pragma unroll
      for (int i = 0; i < 8; i++) aF[8 + i] = bf_from_bits(lA[mrow * LDP + kb + 8 + i]);
    }
    #pragma unroll
    for (int t = 0; t < 4; t++) {
      v16bf bF;
      int col = (t << 4) + (lane & 15);
      int kb  = (lane < 16) ? 0 : 16;
      #pragma unroll
      for (int i = 0; i < 16; i++) bF[i] = bf_from_bits(lB[col * LDP + kb + i]);
      acc[t] = __builtin_amdgcn_wmma_f32_16x16x32_bf16(
          false, aF, false, bF, (short)0, acc[t], false, false);
    }
    __syncthreads();
  }

  // epilogue: bias (+ReLU), f32 or bf16 store
  float*          Cf = (float*)C_;
  unsigned short* Cb = (unsigned short*)C_;
  const int mrow = m0 + (wv << 4) + ((lane < 16) ? 0 : 8);
  #pragma unroll
  for (int t = 0; t < 4; t++) {
    int n = n0 + (t << 4) + (lane & 15);
    if (n >= N) continue;
    float bval = bias ? bias[n] : 0.0f;
    #pragma unroll
    for (int r = 0; r < 8; r++) {
      int m = mrow + r;
      if (m < M) {
        float v = acc[t][r] + bval;
        if (RELU) v = v > 0.0f ? v : 0.0f;
        if (OUT_BF16) Cb[(size_t)m * N + n] = f32_to_bf16_bits(v);
        else          Cf[(size_t)m * N + n] = v;
      }
    }
  }
}

// ---------------- attention (one wave32 per (n, head, q-position)) ----------------
// Q layout [Nseq, Sq, kD]; K/V layout [Nseq, Sk, kD]; lane = key index for the
// score/softmax phase (Sk <= 32), lane = head-dim for the AV phase (kHD == 32).
__global__ __launch_bounds__(128)
void attn_kernel(const float* __restrict__ Q, const float* __restrict__ Kk,
                 const float* __restrict__ Vv, float* __restrict__ O,
                 int Nseq, int Sq, int Sk, const float* __restrict__ addmask)
{
  int widx = blockIdx.x * 4 + (threadIdx.x >> 5);
  int lane = threadIdx.x & 31;
  if (widx >= Nseq * kNH * Sq) return;
  int qs = widx % Sq;
  int h  = (widx / Sq) % kNH;
  int n  = widx / (Sq * kNH);

  const float* qrow = Q + (size_t)(n * Sq + qs) * kD + h * kHD;
  float score = -1e30f;
  if (lane < Sk) {
    const float* krow = Kk + (size_t)(n * Sk + lane) * kD + h * kHD;
    float s = 0.0f;
    #pragma unroll 8
    for (int d = 0; d < kHD; d++) s += qrow[d] * krow[d];
    s *= 0.1767766953f;                       // 1/sqrt(32)
    if (addmask) s += addmask[qs * Sk + lane];
    score = s;
  }
  float m = score;
  for (int o = 16; o > 0; o >>= 1) m = fmaxf(m, __shfl_xor(m, o));
  float e = (lane < Sk) ? __expf(score - m) : 0.0f;
  float ssum = e;
  for (int o = 16; o > 0; o >>= 1) ssum += __shfl_xor(ssum, o);
  float a = e / ssum;

  float outv = 0.0f;                           // lane = head-dim element
  for (int kk = 0; kk < Sk; kk++) {
    float ak = __shfl(a, kk);
    outv += ak * Vv[(size_t)(n * Sk + kk) * kD + h * kHD + lane];
  }
  O[(size_t)(n * Sq + qs) * kD + h * kHD + lane] = outv;
}

// ---------------- residual + LayerNorm (one wave32 per 128-dim row) ----------------
__global__ __launch_bounds__(128)
void add_ln_kernel(const float* __restrict__ X, const float* __restrict__ R,
                   const float* __restrict__ g, const float* __restrict__ b,
                   float* __restrict__ O, int M)
{
  int row  = blockIdx.x * 4 + (threadIdx.x >> 5);
  int lane = threadIdx.x & 31;
  if (row >= M) return;
  float4 xv = ((const float4*)(X + (size_t)row * kD))[lane];
  if (R) {
    float4 rv = ((const float4*)(R + (size_t)row * kD))[lane];
    xv.x += rv.x; xv.y += rv.y; xv.z += rv.z; xv.w += rv.w;
  }
  float s = xv.x + xv.y + xv.z + xv.w;
  for (int o = 16; o > 0; o >>= 1) s += __shfl_xor(s, o);
  float mean = s * (1.0f / kD);
  float dx = xv.x - mean, dy = xv.y - mean, dz = xv.z - mean, dw = xv.w - mean;
  float vs = dx * dx + dy * dy + dz * dz + dw * dw;
  for (int o = 16; o > 0; o >>= 1) vs += __shfl_xor(vs, o);
  float rstd = rsqrtf(vs * (1.0f / kD) + 1e-5f);
  float4 gv = ((const float4*)g)[lane], bv = ((const float4*)b)[lane];
  float4 ov;
  ov.x = dx * rstd * gv.x + bv.x;
  ov.y = dy * rstd * gv.y + bv.y;
  ov.z = dz * rstd * gv.z + bv.z;
  ov.w = dw * rstd * gv.w + bv.w;
  ((float4*)(O + (size_t)row * kD))[lane] = ov;
}

// ---------------- embedding * sqrt(D) + positional encoding ----------------
__global__ void pe_kernel(float* __restrict__ pe)
{
  int idx = blockIdx.x * blockDim.x + threadIdx.x;
  if (idx >= kMAXLEN * (kD / 2)) return;
  int pos = idx / (kD / 2), i = idx % (kD / 2);
  float div = __expf(-logf(10000.0f) * (2.0f * i) / (float)kD);
  float a = (float)pos * div;
  pe[pos * kD + 2 * i]     = __sinf(a);
  pe[pos * kD + 2 * i + 1] = __cosf(a);
}

__global__ void embed_kernel(const int* __restrict__ tok, const float* __restrict__ emb,
                             const float* __restrict__ pe, float* __restrict__ O,
                             int Nrows)
{
  int idx = blockIdx.x * blockDim.x + threadIdx.x;
  if (idx >= Nrows * kD) return;
  int d = idx & (kD - 1);
  int row = idx >> 7;
  int s = row % kS;
  int t = tok[row];
  O[idx] = emb[(size_t)t * kD + d] * 11.3137084989f /* sqrt(128) */ + pe[s * kD + d];
}

// ---------------- history similarity / memory ----------------
// ic[j*kS+s] = dot(cur[j/HIST, s, :], hist[j, s, :])  — one wave per (j,s)
__global__ __launch_bounds__(128)
void simdot_kernel(const float* __restrict__ cur, const float* __restrict__ hist,
                   float* __restrict__ ic)
{
  int widx = blockIdx.x * 4 + (threadIdx.x >> 5);
  int lane = threadIdx.x & 31;
  if (widx >= kB * kHIST * kS) return;
  int s = widx % kS, j = widx / kS, n = j >> 3;
  const float* c = cur  + (size_t)(n * kS + s) * kD;
  const float* h = hist + (size_t)(j * kS + s) * kD;
  float acc = 0.0f;
  #pragma unroll
  for (int i = 0; i < 4; i++) acc += c[lane * 4 + i] * h[lane * 4 + i];
  for (int o = 16; o > 0; o >>= 1) acc += __shfl_xor(acc, o);
  if (lane == 0) ic[j * kS + s] = acc;
}

// per-j: sims_raw[j] = max_s ic[j,s]; then ic[j,:] = softmax over s (in place)
__global__ __launch_bounds__(128)
void icmax_softmax_kernel(float* __restrict__ ic, float* __restrict__ simsraw)
{
  int j    = blockIdx.x * 4 + (threadIdx.x >> 5);
  int lane = threadIdx.x & 31;
  if (j >= kB * kHIST) return;
  float v = (lane < kS) ? ic[j * kS + lane] : -1e30f;
  float m = v;
  for (int o = 16; o > 0; o >>= 1) m = fmaxf(m, __shfl_xor(m, o));
  if (lane == 0) simsraw[j] = m;
  float e = (lane < kS) ? __expf(v - m) : 0.0f;
  float s = e;
  for (int o = 16; o > 0; o >>= 1) s += __shfl_xor(s, o);
  if (lane < kS) ic[j * kS + lane] = e / s;
}

__global__ __launch_bounds__(128)
void sims_softmax_kernel(const float* __restrict__ simsraw, float* __restrict__ sims)
{
  int n    = blockIdx.x * 4 + (threadIdx.x >> 5);
  int lane = threadIdx.x & 31;
  if (n >= kB) return;
  float v = (lane < kHIST) ? simsraw[n * kHIST + lane] : -1e30f;
  float m = v;
  for (int o = 16; o > 0; o >>= 1) m = fmaxf(m, __shfl_xor(m, o));
  float e = (lane < kHIST) ? __expf(v - m) : 0.0f;
  float s = e;
  for (int o = 16; o > 0; o >>= 1) s += __shfl_xor(s, o);
  if (lane < kHIST) sims[n * kHIST + lane] = e / s;
}

// mem[n,d] = sum_h sims[n*H+h] * sum_s ic[j,s] * hist[j,s,d]
__global__ void memsum_kernel(const float* __restrict__ ic, const float* __restrict__ sims,
                              const float* __restrict__ hist, float* __restrict__ memv)
{
  int idx = blockIdx.x * blockDim.x + threadIdx.x;
  if (idx >= kB * kD) return;
  int d = idx & (kD - 1), n = idx >> 7;
  float acc = 0.0f;
  for (int h = 0; h < kHIST; h++) {
    int j = n * kHIST + h;
    float ws = 0.0f;
    for (int s = 0; s < kS; s++)
      ws += ic[j * kS + s] * hist[(size_t)(j * kS + s) * kD + d];
    acc += sims[j] * ws;
  }
  memv[idx] = acc;
}

// batch-major [n*kS+s] -> seq-major [s*kB+n] (reference output order)
__global__ void bs2sb_kernel(const float* __restrict__ X, float* __restrict__ Y)
{
  int idx = blockIdx.x * blockDim.x + threadIdx.x;
  if (idx >= kB * kS * kD) return;
  int d = idx & (kD - 1), r = idx >> 7;
  int s = r % kS, n = r / kS;
  Y[(size_t)(s * kB + n) * kD + d] = X[idx];
}

// ---------------- host orchestration ----------------
extern "C" void kernel_launch(void* const* d_in, const int* in_sizes, int n_in,
                              void* d_out, int out_size, void* d_ws, size_t ws_size,
                              hipStream_t stream)
{
  (void)in_sizes; (void)n_in; (void)out_size; (void)ws_size;
  auto F = [&](int i) { return (const float*)d_in[i]; };
  auto I = [&](int i) { return (const int*)d_in[i]; };

  const int*   src      = I(0);
  const int*   tgt      = I(1);
  const float* tgt_mask = F(2);       // [kS,kS] causal additive mask
  const int*   hist_tok = I(5);
  // inputs 3/4/6 (pad masks, all-false) and 7 (hist_sizes == HIST) unused here.

  // params flattened jax-pytree style (sorted dict keys) after the 8 positional inputs
  const int P = 8;
  const float* b_out    = F(P + 0);
  const int decBase0    = P + 1;               // 3 dec layers x 26 leaves
  const float* dec_ln_b = F(P + 79);
  const float* dec_ln_g = F(P + 80);
  const float* emb      = F(P + 81);
  const int encBase0    = P + 82;              // 3 enc layers x 16 leaves
  const float* enc_ln_b = F(P + 130);
  const float* enc_ln_g = F(P + 131);
  const float* w_out    = F(P + 132);

  // bump allocator over workspace
  char* wsp = (char*)d_ws;
  size_t off = 0;
  auto alloc = [&](size_t bytes) -> void* {
    off = (off + 255) & ~(size_t)255;
    void* p = wsp + off; off += bytes; return p;
  };
  const int Ms = kB * kS;                 // 2432
  const int Mh = kB * kHIST * kS;         // 19456
  float* pe   = (float*)alloc((size_t)kMAXLEN * kD * 4);
  float* xs   = (float*)alloc((size_t)Ms * kD * 4);
  float* xh   = (float*)alloc((size_t)Mh * kD * 4);
  float* qb   = (float*)alloc((size_t)Mh * kD * 4);
  float* kb   = (float*)alloc((size_t)Mh * kD * 4);
  float* vb   = (float*)alloc((size_t)Mh * kD * 4);
  float* ab   = (float*)alloc((size_t)Mh * kD * 4);
  float* tb   = (float*)alloc((size_t)Mh * kD * 4);
  unsigned short* hb = (unsigned short*)alloc((size_t)Mh * kFF * 2);  // bf16 FFN hidden
  float* ic      = (float*)alloc((size_t)kB * kHIST * kS * 4);
  float* simsraw = (float*)alloc((size_t)kB * kHIST * 4);
  float* sims    = (float*)alloc((size_t)kB * kHIST * 4);
  float* memv    = (float*)alloc((size_t)kB * kD * 4);
  float* kmem    = (float*)alloc((size_t)kB * kD * 4);
  float* vmem    = (float*)alloc((size_t)kB * kD * 4);
  float* yb      = (float*)alloc((size_t)Ms * kD * 4);
  float* ysb     = (float*)alloc((size_t)Ms * kD * 4);

  // attn weight pointers in sorted-key order {bk,bo,bq,bv,wk,wo,wq,wv} -> (wq,bq,wk,bk,wv,bv,wo,bo)
  auto attnW = [&](int base, const float** w) {
    w[0] = F(base + 6); w[1] = F(base + 2);   // wq, bq
    w[2] = F(base + 4); w[3] = F(base + 0);   // wk, bk
    w[4] = F(base + 7); w[5] = F(base + 3);   // wv, bv
    w[6] = F(base + 5); w[7] = F(base + 1);   // wo, bo
  };
  auto gemm = [&](const void* A, const float* W, const float* bias, void* C,
                  int M, int N, int K, int mode) {
    dim3 g((N + 63) / 64, (M + 63) / 64), blk(128);
    if (mode == 0)      gemm_wmma<false, false, false><<<g, blk, 0, stream>>>(A, W, bias, C, M, N, K);
    else if (mode == 1) gemm_wmma<false, true,  true ><<<g, blk, 0, stream>>>(A, W, bias, C, M, N, K);
    else                gemm_wmma<true,  false, false><<<g, blk, 0, stream>>>(A, W, bias, C, M, N, K);
  };
  auto attn = [&](const float* Q, const float* Kp, const float* V, float* O,
                  int Nseq, int Sq, int Sk, const float* msk) {
    int nw = Nseq * kNH * Sq;
    attn_kernel<<<(nw + 3) / 4, 128, 0, stream>>>(Q, Kp, V, O, Nseq, Sq, Sk, msk);
  };
  auto ln = [&](const float* X, const float* R, const float* g, const float* b,
                float* O, int M) {
    add_ln_kernel<<<(M + 3) / 4, 128, 0, stream>>>(X, R, g, b, O, M);
  };
  auto enc_layer = [&](float* x, int Nseq, int base) {
    int M = Nseq * kS;
    const float* w[8]; attnW(base + 6, w);
    gemm(x, w[0], w[1], qb, M, kD, kD, 0);
    gemm(x, w[2], w[3], kb, M, kD, kD, 0);
    gemm(x, w[4], w[5], vb, M, kD, kD, 0);
    attn(qb, kb, vb, ab, Nseq, kS, kS, nullptr);
    gemm(ab, w[6], w[7], tb, M, kD, kD, 0);
    ln(x, tb, F(base + 3), F(base + 2), x, M);                 // ln1
    gemm(x,  F(base + 14), F(base + 0), hb, M, kFF, kD, 1);    // relu -> bf16
    gemm(hb, F(base + 15), F(base + 1), tb, M, kD, kFF, 2);    // bf16 A
    ln(x, tb, F(base + 5), F(base + 4), x, M);                 // ln2
  };

  // positional encodings + embeddings
  pe_kernel<<<(kMAXLEN * (kD / 2) + 127) / 128, 128, 0, stream>>>(pe);
  embed_kernel<<<(Ms * kD + 255) / 256, 256, 0, stream>>>(src, emb, pe, xs, Ms);

  // encode current sessions
  for (int l = 0; l < kNL; l++) enc_layer(xs, kB, encBase0 + 16 * l);
  ln(xs, nullptr, enc_ln_g, enc_ln_b, xs, Ms);

  // encode history sessions
  embed_kernel<<<(Mh * kD + 255) / 256, 256, 0, stream>>>(hist_tok, emb, pe, xh, Mh);
  for (int l = 0; l < kNL; l++) enc_layer(xh, kB * kHIST, encBase0 + 16 * l);
  ln(xh, nullptr, enc_ln_g, enc_ln_b, xh, Mh);

  // history similarity -> decoder memory [kB, kD]
  simdot_kernel<<<(kB * kHIST * kS + 3) / 4, 128, 0, stream>>>(xs, xh, ic);
  icmax_softmax_kernel<<<(kB * kHIST + 3) / 4, 128, 0, stream>>>(ic, simsraw);
  sims_softmax_kernel<<<(kB + 3) / 4, 128, 0, stream>>>(simsraw, sims);
  memsum_kernel<<<(kB * kD + 255) / 256, 256, 0, stream>>>(ic, sims, xh, memv);

  // decoder
  embed_kernel<<<(Ms * kD + 255) / 256, 256, 0, stream>>>(tgt, emb, pe, yb, Ms);
  for (int l = 0; l < kNL; l++) {
    int base = decBase0 + 26 * l;
    const float* sw[8]; attnW(base + 16, sw);   // self-attn
    gemm(yb, sw[0], sw[1], qb, Ms, kD, kD, 0);
    gemm(yb, sw[2], sw[3], kb, Ms, kD, kD, 0);
    gemm(yb, sw[4], sw[5], vb, Ms, kD, kD, 0);
    attn(qb, kb, vb, ab, kB, kS, kS, tgt_mask);
    gemm(ab, sw[6], sw[7], tb, Ms, kD, kD, 0);
    ln(yb, tb, F(base + 11), F(base + 10), yb, Ms);            // ln1

    const float* cw[8]; attnW(base + 2, cw);    // cross-attn (memory len 1)
    gemm(yb,   cw[0], cw[1], qb,   Ms, kD, kD, 0);
    gemm(memv, cw[2], cw[3], kmem, kB, kD, kD, 0);
    gemm(memv, cw[4], cw[5], vmem, kB, kD, kD, 0);
    attn(qb, kmem, vmem, ab, kB, kS, 1, nullptr);
    gemm(ab, cw[6], cw[7], tb, Ms, kD, kD, 0);
    ln(yb, tb, F(base + 13), F(base + 12), yb, Ms);            // ln2

    gemm(yb, F(base + 24), F(base + 0), hb, Ms, kFF, kD, 1);   // relu -> bf16
    gemm(hb, F(base + 25), F(base + 1), tb, Ms, kD, kFF, 2);   // bf16 A
    ln(yb, tb, F(base + 15), F(base + 14), yb, Ms);            // ln3
  }
  ln(yb, nullptr, dec_ln_g, dec_ln_b, yb, Ms);

  // reorder to [S,B,D] then vocab projection -> d_out [S,B,VOCAB] f32
  bs2sb_kernel<<<(Ms * kD + 255) / 256, 256, 0, stream>>>(yb, ysb);
  gemm(ysb, w_out, b_out, (float*)d_out, Ms, kVOCAB, kD, 0);
}
